// Decoder_19774029431790
// MI455X (gfx1250) — matrix operand order repaired and checked by reading, not compiled, not run on hardware
//
#include <hip/hip_runtime.h>
#include <cstdint>
#include <cstddef>

typedef __attribute__((ext_vector_type(16))) _Float16 v16h;
typedef __attribute__((ext_vector_type(8)))  _Float16 v8h;
typedef __attribute__((ext_vector_type(8)))  float    v8f;

union H16 { v16h v; v8h q[2]; _Float16 h[16]; };

// ---------------- Stage A: channel mean / max over 888 channels ----------------
__global__ void k_reduce(const float* __restrict__ x, float* __restrict__ pad1,
                         float* __restrict__ pad2) {
  int idx = blockIdx.x * 256 + threadIdx.x;            // 8192 = 8 * 1024
  int n = idx >> 10, p = idx & 1023;
  const float* xp = x + (size_t)n * 888 * 1024 + p;
  float s = 0.f, mx = -3.4e38f;
  for (int c = 0; c < 888; ++c) { float v = xp[(size_t)c * 1024]; s += v; mx = fmaxf(mx, v); }
  pad1[idx] = s * (1.0f / 888.0f);
  pad2[idx] = mx;
}

// ---------------- Weight prepack (f32 -> f16, K padded) ----------------
__global__ void k_packw1(const float* __restrict__ W1, _Float16* __restrict__ wh) {
  int i = blockIdx.x * 256 + threadIdx.x;              // 64 * 896
  if (i >= 64 * 896) return;
  int co = i / 896, c = i % 896;
  wh[i] = (c < 888) ? (_Float16)W1[co * 888 + c] : (_Float16)0.f;
}
__global__ void k_packwl(const float* __restrict__ Wl, _Float16* __restrict__ whl) {
  int i = blockIdx.x * 256 + threadIdx.x;              // 9 * 64 * 96
  if (i >= 9 * 64 * 96) return;
  int c = i % 96; int rest = i / 96; int co = rest % 64; int tap = rest / 64;
  // Wlast layout [64][66][3][3]; tap = dy*3+dx
  whl[i] = (c < 66) ? (_Float16)Wl[(co * 66 + c) * 9 + tap] : (_Float16)0.f;
}

// ---------------- Stage B: 1x1 conv 888 -> 64 via WMMA f16 ----------------
// wave = 16 positions (M) x 64 out-channels (4 N-tiles), K = 896 (28 steps of 32)
__global__ void k_gemm1x1(const float* __restrict__ x, const _Float16* __restrict__ wh,
                          const float* __restrict__ b1, float* __restrict__ x64) {
  int lane = threadIdx.x & 31;
  int wave = threadIdx.x >> 5;
  int mt = blockIdx.x * 4 + wave;                      // 0..511
  int n = mt >> 6;
  int pbase = (mt & 63) << 4;
  int half = lane >> 4;
  int m = lane & 15;
  int p = pbase + m;
  const float* xn = x + (size_t)n * 888 * 1024;
  v8f vz = {};
  v8f acc[4]; acc[0] = vz; acc[1] = vz; acc[2] = vz; acc[3] = vz;

  for (int ks = 0; ks < 28; ++ks) {
    int kb = ks * 32;
    int ka  = kb + half * 8;       // A rows: h[0..7]  = K = ka..ka+7
    int kaa = kb + 16 + half * 8;  //         h[8..15] = K = kaa..kaa+7
    H16 A;
    if (kb + 32 <= 888) {
#pragma unroll
      for (int j = 0; j < 8; ++j) A.h[j]     = (_Float16)xn[(ka  + j) * 1024 + p];
#pragma unroll
      for (int j = 0; j < 8; ++j) A.h[8 + j] = (_Float16)xn[(kaa + j) * 1024 + p];
    } else {                        // tail: channels >= 888 are zero
#pragma unroll
      for (int j = 0; j < 8; ++j) A.h[j]     = (ka  + j < 888) ? (_Float16)xn[(ka  + j) * 1024 + p] : (_Float16)0.f;
#pragma unroll
      for (int j = 0; j < 8; ++j) A.h[8 + j] = (kaa + j < 888) ? (_Float16)xn[(kaa + j) * 1024 + p] : (_Float16)0.f;
    }
#pragma unroll
    for (int nt = 0; nt < 4; ++nt) {
      H16 Bm;                       // column-per-lane: h[j] = K = kb + 8*half + j, h[8+j] = K = kb + 16 + 8*half + j
      const _Float16* bb = wh + (size_t)(nt * 16 + m) * 896 + kb + half * 8;
      Bm.q[0] = *(const v8h*)bb;
      Bm.q[1] = *(const v8h*)(bb + 16);
      acc[nt] = __builtin_amdgcn_wmma_f32_16x16x32_f16(false, A.v, false, Bm.v,
                                                       (short)0, acc[nt], false, false);
    }
  }
#pragma unroll
  for (int nt = 0; nt < 4; ++nt) {
    int co = nt * 16 + m;
    float bias = b1[co];
    float* o = x64 + ((size_t)(n * 64 + co)) * 1024 + pbase + half * 8; // M = r + 8*half
#pragma unroll
    for (int r = 0; r < 8; ++r) o[r] = acc[nt][r] + bias;
  }
}

// ---------------- Stage C1: shuffle + depthwise 3/5/7 fusion ----------------
__global__ void k_fuse(const float* __restrict__ x64, const float* __restrict__ pad1,
                       const float* __restrict__ pad2,
                       const float* __restrict__ W3, const float* __restrict__ b3,
                       const float* __restrict__ W5, const float* __restrict__ b5,
                       const float* __restrict__ W7, const float* __restrict__ b7,
                       float* __restrict__ xf) {
  int idx = blockIdx.x * 256 + threadIdx.x;            // 8*66*1024
  if (idx >= 8 * 66 * 1024) return;
  int p = idx & 1023;
  int t = idx >> 10;
  int fc = t % 66, n = t / 66;
  int g = fc / 22, c = fc % 22;
  int s = 3 * c + g;                                   // shuffled source channel
  const float* Wg = (g == 0) ? W3 : (g == 1) ? W5 : W7;
  const float* bg = (g == 0) ? b3 : (g == 1) ? b5 : b7;
  int ksz = 3 + 2 * g;
  int pad = 1 + g;
  int h = p >> 5, w = p & 31;
  const float* src = (s < 64) ? (x64 + ((size_t)(n * 64 + s)) * 1024)
                   : (s == 64) ? (pad1 + n * 1024) : (pad2 + n * 1024);
  float acc = bg[c];
  const float* wk = Wg + c * ksz * ksz;
  for (int ky = 0; ky < ksz; ++ky) {
    int hh = h + ky - pad;
    if ((unsigned)hh >= 32u) continue;
    for (int kx = 0; kx < ksz; ++kx) {
      int ww = w + kx - pad;
      if ((unsigned)ww >= 32u) continue;
      acc += wk[ky * ksz + kx] * src[hh * 32 + ww];
    }
  }
  xf[idx] = acc;
}

// ---------------- Stage C2: PE dwconv + residual + LayerNorm(C) -> f16 NHWC pad96 ----------------
__global__ void k_peln(const float* __restrict__ xf, const float* __restrict__ Wpe,
                       const float* __restrict__ bpe, const float* __restrict__ lg,
                       const float* __restrict__ lb, float* __restrict__ xf2,
                       _Float16* __restrict__ nh) {
  int idx = blockIdx.x * 256 + threadIdx.x;            // 8192
  int n = idx >> 10, p = idx & 1023;
  int h = p >> 5, w = p & 31;
  const float* xn = xf + (size_t)n * 66 * 1024;
  float s1 = 0.f, s2 = 0.f;
  for (int c = 0; c < 66; ++c) {
    const float* plane = xn + (size_t)c * 1024;
    float pe = bpe[c];
    const float* wk = Wpe + c * 9;
#pragma unroll
    for (int ky = 0; ky < 3; ++ky) {
      int hh = h + ky - 1;
      if ((unsigned)hh >= 32u) continue;
#pragma unroll
      for (int kx = 0; kx < 3; ++kx) {
        int ww = w + kx - 1;
        if ((unsigned)ww >= 32u) continue;
        pe += wk[ky * 3 + kx] * plane[hh * 32 + ww];
      }
    }
    float v = plane[p] + pe;
    xf2[(size_t)(n * 66 + c) * 1024 + p] = v;
    s1 += v; s2 += v * v;
  }
  float mu = s1 * (1.f / 66.f);
  float var = s2 * (1.f / 66.f) - mu * mu;
  float rs = rsqrtf(var + 1e-5f);
  _Float16* o = nh + (size_t)idx * 96;
  for (int c = 0; c < 66; ++c) {
    float v = xf2[(size_t)(n * 66 + c) * 1024 + p];
    o[c] = (_Float16)((v - mu) * rs * lg[c] + lb[c]);
  }
#pragma unroll
  for (int c = 66; c < 96; ++c) o[c] = (_Float16)0.f;
}

// ---------------- Stage D: 3x3 conv 66->64 via WMMA (9 taps x K=96) + GELU + residual ----------------
__global__ void k_wlast(const _Float16* __restrict__ nh, const _Float16* __restrict__ whl,
                        const float* __restrict__ blast, const float* __restrict__ x64,
                        float* __restrict__ outb) {
  int lane = threadIdx.x & 31;
  int wave = threadIdx.x >> 5;
  int mt = blockIdx.x * 4 + wave;                      // 0..511
  int n = mt >> 6;
  int pbase = (mt & 63) << 4;
  int half = lane >> 4;
  int m = lane & 15;
  int p = pbase + m;
  int h0 = p >> 5, w0 = p & 31;
  v8f vz = {};
  v8f acc[4]; acc[0] = vz; acc[1] = vz; acc[2] = vz; acc[3] = vz;
  v8h hz = {};
  for (int tap = 0; tap < 9; ++tap) {
    int dy = tap / 3 - 1, dx = tap % 3 - 1;
    int hh = h0 + dy, ww = w0 + dx;
    bool valid = ((unsigned)hh < 32u) && ((unsigned)ww < 32u);
    int q = valid ? (hh * 32 + ww) : 0;
    const _Float16* ab = nh + ((size_t)(n * 1024 + q)) * 96;
#pragma unroll
    for (int ks = 0; ks < 3; ++ks) {
      int kb = ks * 32;
      H16 A;
      if (valid) {
        A.q[0] = *(const v8h*)(ab + kb + half * 8);
        A.q[1] = *(const v8h*)(ab + kb + 16 + half * 8);
      } else { A.q[0] = hz; A.q[1] = hz; }             // EXEC stays all-ones; zero data
#pragma unroll
      for (int nt = 0; nt < 4; ++nt) {
        H16 Bm;
        const _Float16* bb = whl + ((size_t)(tap * 64 + nt * 16 + m)) * 96 + kb + half * 8;
        Bm.q[0] = *(const v8h*)bb;
        Bm.q[1] = *(const v8h*)(bb + 16);
        acc[nt] = __builtin_amdgcn_wmma_f32_16x16x32_f16(false, A.v, false, Bm.v,
                                                         (short)0, acc[nt], false, false);
      }
    }
  }
#pragma unroll
  for (int nt = 0; nt < 4; ++nt) {
    int co = nt * 16 + m;
    float bias = blast[co];
    size_t base = ((size_t)(n * 64 + co)) * 1024 + pbase + half * 8;
#pragma unroll
    for (int r = 0; r < 8; ++r) {
      float v = acc[nt][r] + bias;
      float gel = 0.5f * v * (1.0f + erff(v * 0.70710678118654752f));  // exact GELU
      outb[base + r] = gel + x64[base + r];
    }
  }
}

// ---------------- Stage E: aux head depthwise convs + 1x1 -> 6 ----------------
__global__ void k_dw3(const float* __restrict__ in, const float* __restrict__ W,
                      float* __restrict__ out) {
  int idx = blockIdx.x * 256 + threadIdx.x;
  if (idx >= 8 * 64 * 1024) return;
  int p = idx & 1023; int t = idx >> 10; int c = t & 63;
  int h = p >> 5, w = p & 31;
  const float* src = in + (size_t)t * 1024;
  const float* wk = W + c * 9;
  float acc = 0.f;
#pragma unroll
  for (int ky = 0; ky < 3; ++ky) {
    int hh = h + ky - 1;
    if ((unsigned)hh >= 32u) continue;
#pragma unroll
    for (int kx = 0; kx < 3; ++kx) {
      int ww = w + kx - 1;
      if ((unsigned)ww >= 32u) continue;
      acc += wk[ky * 3 + kx] * src[hh * 32 + ww];
    }
  }
  out[idx] = acc;
}
__global__ void k_dw5d3(const float* __restrict__ in, const float* __restrict__ W,
                        float* __restrict__ out) {
  int idx = blockIdx.x * 256 + threadIdx.x;
  if (idx >= 8 * 64 * 1024) return;
  int p = idx & 1023; int t = idx >> 10; int c = t & 63;
  int h = p >> 5, w = p & 31;
  const float* src = in + (size_t)t * 1024;
  const float* wk = W + c * 25;
  float acc = 0.f;
#pragma unroll
  for (int ky = 0; ky < 5; ++ky) {
    int hh = h + (ky - 2) * 3;                         // dilation 3, pad 6
    if ((unsigned)hh >= 32u) continue;
#pragma unroll
    for (int kx = 0; kx < 5; ++kx) {
      int ww = w + (kx - 2) * 3;
      if ((unsigned)ww >= 32u) continue;
      acc += wk[ky * 5 + kx] * src[hh * 32 + ww];
    }
  }
  out[idx] = acc;
}
__global__ void k_proj6(const float* __restrict__ in, const float* __restrict__ Wo,
                        float* __restrict__ f6) {
  int idx = blockIdx.x * 256 + threadIdx.x;            // 8192
  int n = idx >> 10, p = idx & 1023;
  float a[6] = {0.f, 0.f, 0.f, 0.f, 0.f, 0.f};
  for (int c = 0; c < 64; ++c) {
    float v = in[((size_t)(n * 64 + c)) * 1024 + p];
#pragma unroll
    for (int o = 0; o < 6; ++o) a[o] += Wo[o * 64 + c] * v;
  }
#pragma unroll
  for (int o = 0; o < 6; ++o) f6[((size_t)(n * 6 + o)) * 1024 + p] = a[o];
}

// ---------------- Stage F: bilinear upsample 32x32 -> 1024x1024 (half-pixel, clamp) ----------------
__global__ void k_upsample(const float* __restrict__ f6, float* __restrict__ out, int total) {
  int o = blockIdx.x * 256 + threadIdx.x;
  if (o >= total) return;
  int xo = o & 1023;
  int yo = (o >> 10) & 1023;
  int nc = o >> 20;                                    // n*6 + ch
  float sx = (xo + 0.5f) * (1.0f / 32.0f) - 0.5f;
  float sy = (yo + 0.5f) * (1.0f / 32.0f) - 0.5f;
  int x0 = (int)floorf(sx); float fx = sx - (float)x0;
  int y0 = (int)floorf(sy); float fy = sy - (float)y0;
  int x1 = min(max(x0 + 1, 0), 31); x0 = min(max(x0, 0), 31);
  int y1 = min(max(y0 + 1, 0), 31); y0 = min(max(y0, 0), 31);
  const float* pl = f6 + (size_t)nc * 1024;
  float v00 = pl[y0 * 32 + x0], v01 = pl[y0 * 32 + x1];
  float v10 = pl[y1 * 32 + x0], v11 = pl[y1 * 32 + x1];
  float v0 = v00 + (v01 - v00) * fx;
  float v1 = v10 + (v11 - v10) * fx;
  out[o] = v0 + (v1 - v0) * fy;
}

extern "C" void kernel_launch(void* const* d_in, const int* in_sizes, int n_in,
                              void* d_out, int out_size, void* d_ws, size_t ws_size,
                              hipStream_t stream) {
  const float* x   = (const float*)d_in[0];
  const float* W1  = (const float*)d_in[3];
  const float* b1  = (const float*)d_in[4];
  const float* W3  = (const float*)d_in[5];
  const float* b3  = (const float*)d_in[6];
  const float* W5  = (const float*)d_in[7];
  const float* b5  = (const float*)d_in[8];
  const float* W7  = (const float*)d_in[9];
  const float* b7  = (const float*)d_in[10];
  const float* Wpe = (const float*)d_in[11];
  const float* bpe = (const float*)d_in[12];
  const float* lg  = (const float*)d_in[13];
  const float* lb  = (const float*)d_in[14];
  const float* Wl  = (const float*)d_in[15];
  const float* bl  = (const float*)d_in[16];
  const float* Wa0 = (const float*)d_in[17];
  const float* Was = (const float*)d_in[18];
  const float* Wo  = (const float*)d_in[19];
  float* out = (float*)d_out;

  char* ws = (char*)d_ws;
  size_t off = 0;
  auto alloc = [&](size_t bytes) -> void* {
    off = (off + 255) & ~(size_t)255;
    void* p = ws + off;
    off += bytes;
    return p;
  };
  float*    x64  = (float*)   alloc((size_t)8 * 64 * 1024 * 4);
  float*    pad1 = (float*)   alloc((size_t)8 * 1024 * 4);
  float*    pad2 = (float*)   alloc((size_t)8 * 1024 * 4);
  _Float16* wh   = (_Float16*)alloc((size_t)64 * 896 * 2);
  _Float16* whl  = (_Float16*)alloc((size_t)9 * 64 * 96 * 2);
  float*    xf   = (float*)   alloc((size_t)8 * 66 * 1024 * 4);
  float*    xf2  = (float*)   alloc((size_t)8 * 66 * 1024 * 4);
  _Float16* nh   = (_Float16*)alloc((size_t)8 * 1024 * 96 * 2);
  float*    outb = (float*)   alloc((size_t)8 * 64 * 1024 * 4);
  float*    f6   = (float*)   alloc((size_t)8 * 6 * 1024 * 4);
  float* feat1 = xf;   // xf dead after k_peln -> reuse for aux head
  float* feat2 = xf2;  // xf2 dead after k_peln -> reuse

  k_reduce  <<<32, 256, 0, stream>>>(x, pad1, pad2);
  k_packw1  <<<(64 * 896 + 255) / 256, 256, 0, stream>>>(W1, wh);
  k_packwl  <<<(9 * 64 * 96 + 255) / 256, 256, 0, stream>>>(Wl, whl);
  k_gemm1x1 <<<128, 128, 0, stream>>>(x, wh, b1, x64);
  k_fuse    <<<(8 * 66 * 1024) / 256, 256, 0, stream>>>(x64, pad1, pad2, W3, b3, W5, b5, W7, b7, xf);
  k_peln    <<<32, 256, 0, stream>>>(xf, Wpe, bpe, lg, lb, xf2, nh);
  k_wlast   <<<128, 128, 0, stream>>>(nh, whl, bl, x64, outb);
  k_dw3     <<<(8 * 64 * 1024) / 256, 256, 0, stream>>>(outb, Wa0, feat1);
  k_dw5d3   <<<(8 * 64 * 1024) / 256, 256, 0, stream>>>(feat1, Was, feat2);
  k_proj6   <<<32, 256, 0, stream>>>(feat2, Wo, f6);
  int total = 8 * 6 * 1024 * 1024;
  k_upsample<<<total / 256, 256, 0, stream>>>(f6, out, total);
}
